// VQVAE_EMA_43026982372009
// MI455X (gfx1250) — matrix operand order repaired and checked by reading, not compiled, not run on hardware
//
#include <hip/hip_runtime.h>
#include <hip/hip_bf16.h>
#include <math.h>

typedef __attribute__((ext_vector_type(16))) __bf16       v16bf;
typedef __attribute__((ext_vector_type(8)))  float        v8f;
typedef __attribute__((ext_vector_type(4)))  unsigned int v4u;
typedef __attribute__((ext_vector_type(4)))  float        v4f;

#define B_TOT   262144
#define IN_DIM  225
#define LATENT  16
#define OUT_DIM 45
#define NUM_EMB 512
#define PROJ    8

#define BLOCK_THREADS 256
#define WAVES_PER_BLK 8
#define GRID_BLKS     512

// LDS row strides (elements); +8 bf16 pad breaks bank alignment
#define S_W1  264
#define S_W2  136
#define S_WMU 72
#define S_EP  40
#define S_WD1 40
#define S_WD2 72
#define S_WD3 136
#define S_STG 136
#define S_ZST 17
#define S_ZP  9

// ---- dynamic-LDS byte offsets (all 16B aligned) ----
constexpr unsigned OFF_W1  = 0;
constexpr unsigned OFF_W2  = OFF_W1  + 128u*S_W1*2;   // 67584
constexpr unsigned OFF_WMU = OFF_W2  + 64u*S_W2*2;    // +17408
constexpr unsigned OFF_EP  = OFF_WMU + 16u*S_WMU*2;   // +2304
constexpr unsigned OFF_ES  = OFF_EP  + 512u*S_EP*2;   // +40960
constexpr unsigned OFF_EMB = OFF_ES  + 512u*4;        // +2048
constexpr unsigned OFF_WD1 = OFF_EMB + 512u*16*4;     // +32768
constexpr unsigned OFF_WD2 = OFF_WD1 + 64u*S_WD1*2;   // +5120
constexpr unsigned OFF_WD3 = OFF_WD2 + 128u*S_WD2*2;  // +18432
constexpr unsigned OFF_WP  = OFF_WD3 + 48u*S_WD3*2;   // +13056
constexpr unsigned OFF_BE1 = OFF_WP  + 8u*16*4;       // +512
constexpr unsigned OFF_BE2 = OFF_BE1 + 128u*4;
constexpr unsigned OFF_BMU = OFF_BE2 + 64u*4;
constexpr unsigned OFF_BD1 = OFF_BMU + 16u*4;
constexpr unsigned OFF_BD2 = OFF_BD1 + 64u*4;
constexpr unsigned OFF_BD3 = OFF_BD2 + 128u*4;
constexpr unsigned OFF_BP  = OFF_BD3 + 48u*4;
constexpr unsigned OFF_STG = OFF_BP  + 8u*4;
constexpr unsigned OFF_ZST = OFF_STG + 8u*16*S_STG*2; // +34816
constexpr unsigned OFF_ZP  = OFF_ZST + 8u*16*S_ZST*4; // +8704
constexpr unsigned OFF_CLO = OFF_ZP  + 8u*16*S_ZP*4;  // +4608
constexpr unsigned LDS_TOTAL = OFF_CLO + 8u*16*4;     // ~250 KB < 320 KB

// ---- fragment loaders (ISA 7.12.2 layouts, wave32) ----
// A 16x32 bf16: lane L -> row = L&15; elems 0..7  : K = (L>>4)*8 + i
//                                  elems 8..15 : K = 16 + (L>>4)*8 + (i-8)
// p = rowbase + kc*32 + grp*8  (16B aligned)
__device__ __forceinline__ v16bf ld_afrag(const __bf16* p) {
  union { v16bf v; v4u u[2]; } r;
  r.u[0] = *(const v4u*)(p);
  r.u[1] = *(const v4u*)(p + 16);
  return r.v;
}
// B 32x16 bf16: lane L -> col = L&15; elems 0..7 : K = (L>>4)*8 + i; elems 8..15 : K = 16 + (L>>4)*8 + (i-8)
// p = rowbase(n) + kc*32 + grp*8  (16B aligned)
__device__ __forceinline__ v16bf ld_bfrag(const __bf16* p) {
  union { v16bf v; v4u u[2]; } r;
  r.u[0] = *(const v4u*)(p);
  r.u[1] = *(const v4u*)(p + 16);
  return r.v;
}
__device__ __forceinline__ float eluf(float x) { return x > 0.f ? x : expm1f(x); }

#define WMMA_BF16(A, Bm, C) \
  __builtin_amdgcn_wmma_f32_16x16x32_bf16(false, (A), false, (Bm), (short)0, (C), false, false)

__global__ void __launch_bounds__(BLOCK_THREADS)
vqvae_fused(const float* __restrict__ x,
            const float* __restrict__ w_e1, const float* __restrict__ b_e1,
            const float* __restrict__ w_e2, const float* __restrict__ b_e2,
            const float* __restrict__ w_mu, const float* __restrict__ b_mu,
            const float* __restrict__ w_p,  const float* __restrict__ b_p,
            const float* __restrict__ emb,
            const float* __restrict__ w_d1, const float* __restrict__ b_d1,
            const float* __restrict__ w_d2, const float* __restrict__ b_d2,
            const float* __restrict__ w_d3, const float* __restrict__ b_d3,
            float* __restrict__ o_xhat, float* __restrict__ o_q,
            float* __restrict__ o_z,    float* __restrict__ o_oh)
{
  extern __shared__ __align__(16) char smem[];
  __bf16* lw1  = (__bf16*)(smem + OFF_W1);
  __bf16* lw2  = (__bf16*)(smem + OFF_W2);
  __bf16* lwmu = (__bf16*)(smem + OFF_WMU);
  __bf16* lep  = (__bf16*)(smem + OFF_EP);   // -2*e_p padded to K=32
  float*  les  = (float*)(smem + OFF_ES);    // ||e_p||^2
  float*  lemb = (float*)(smem + OFF_EMB);
  __bf16* lwd1 = (__bf16*)(smem + OFF_WD1);
  __bf16* lwd2 = (__bf16*)(smem + OFF_WD2);
  __bf16* lwd3 = (__bf16*)(smem + OFF_WD3);
  float*  lwp  = (float*)(smem + OFF_WP);
  float*  lbe1 = (float*)(smem + OFF_BE1);
  float*  lbe2 = (float*)(smem + OFF_BE2);
  float*  lbmu = (float*)(smem + OFF_BMU);
  float*  lbd1 = (float*)(smem + OFF_BD1);
  float*  lbd2 = (float*)(smem + OFF_BD2);
  float*  lbd3 = (float*)(smem + OFF_BD3);
  float*  lbp  = (float*)(smem + OFF_BP);
  __bf16* lstg = (__bf16*)(smem + OFF_STG);
  float*  lzst = (float*)(smem + OFF_ZST);
  float*  lzp  = (float*)(smem + OFF_ZP);
  int*    lclo = (int*)(smem + OFF_CLO);

  const int t = threadIdx.x;

  // ================= setup: weights -> bf16 LDS (once per block) =================
  for (int i = t; i < 128 * 256; i += BLOCK_THREADS) {
    int n = i >> 8, k = i & 255;
    lw1[n * S_W1 + k] = (k < IN_DIM) ? (__bf16)w_e1[n * IN_DIM + k] : (__bf16)0.f;
  }
  for (int i = t; i < 64 * 128; i += BLOCK_THREADS) {
    int n = i >> 7, k = i & 127;
    lw2[n * S_W2 + k] = (__bf16)w_e2[n * 128 + k];
  }
  for (int i = t; i < 16 * 64; i += BLOCK_THREADS) {
    int n = i >> 6, k = i & 63;
    lwmu[n * S_WMU + k] = (__bf16)w_mu[n * 64 + k];
  }
  for (int i = t; i < 64 * 32; i += BLOCK_THREADS) {
    int n = i >> 5, k = i & 31;
    lwd1[n * S_WD1 + k] = (k < LATENT) ? (__bf16)w_d1[n * LATENT + k] : (__bf16)0.f;
  }
  for (int i = t; i < 128 * 64; i += BLOCK_THREADS) {
    int n = i >> 6, k = i & 63;
    lwd2[n * S_WD2 + k] = (__bf16)w_d2[n * 64 + k];
  }
  for (int i = t; i < 48 * 128; i += BLOCK_THREADS) {
    int n = i >> 7, k = i & 127;
    lwd3[n * S_WD3 + k] = (n < OUT_DIM) ? (__bf16)w_d3[n * 128 + k] : (__bf16)0.f;
  }
  for (int i = t; i < PROJ * LATENT; i += BLOCK_THREADS) lwp[i] = w_p[i];
  for (int i = t; i < NUM_EMB * LATENT; i += BLOCK_THREADS) lemb[i] = emb[i];
  if (t < 128) lbe1[t] = b_e1[t];
  if (t < 64)  lbe2[t] = b_e2[t];
  if (t < 16)  lbmu[t] = b_mu[t];
  if (t < 64)  lbd1[t] = b_d1[t];
  if (t < 128) lbd2[t] = b_d2[t];
  if (t < 48)  lbd3[t] = (t < OUT_DIM) ? b_d3[t] : 0.f;
  if (t < 8)   lbp[t]  = b_p[t];
  // e_p = emb @ w_p^T + b_p ; store -2*e_p (bf16, K padded to 32) and ||e_p||^2
  for (int n = t; n < NUM_EMB; n += BLOCK_THREADS) {
    float ep[PROJ];
    float ss = 0.f;
#pragma unroll
    for (int k = 0; k < PROJ; ++k) {
      float s = b_p[k];
#pragma unroll
      for (int j = 0; j < LATENT; ++j) s += emb[n * LATENT + j] * w_p[k * LATENT + j];
      ep[k] = s;
      ss += s * s;
    }
    les[n] = ss;
#pragma unroll
    for (int k = 0; k < PROJ; ++k) lep[n * S_EP + k] = (__bf16)(-2.f * ep[k]);
#pragma unroll
    for (int k = PROJ; k < 32; ++k) lep[n * S_EP + k] = (__bf16)0.f;
  }
  __syncthreads();

  // ================= per-wave 16-row tiles =================
  const int wave = t >> 5, lane = t & 31;
  const int nl = lane & 15, grp = lane >> 4;
  __bf16* stg  = lstg + wave * 16 * S_STG;
  float*  zstw = lzst + wave * 16 * S_ZST;
  float*  zpw  = lzp  + wave * 16 * S_ZP;
  int*    clw  = lclo + wave * 16;

  const int numTiles = B_TOT / 16;
  for (int tile = blockIdx.x * WAVES_PER_BLK + wave; tile < numTiles;
       tile += gridDim.x * WAVES_PER_BLK) {
    const int R = tile * 16;
    const float* xr = x + (size_t)(R + nl) * IN_DIM;

    // ---- load x A-fragments (K padded 225 -> 256), NT f32 -> bf16 in regs ----
    v16bf a1[8];
#pragma unroll
    for (int kc = 0; kc < 7; ++kc) {
      int b0 = kc * 32 + grp * 8;
      v16bf a;
#pragma unroll
      for (int i = 0; i < 8; ++i) a[i] = (__bf16)__builtin_nontemporal_load(xr + b0 + i);
#pragma unroll
      for (int i = 0; i < 8; ++i) a[8 + i] = (__bf16)__builtin_nontemporal_load(xr + b0 + 16 + i);
      a1[kc] = a;
    }
    {
      int b0 = 224 + grp * 8;
      v16bf a;
#pragma unroll
      for (int i = 0; i < 8; ++i) {
        int k = b0 + i;
        a[i] = (k < IN_DIM) ? (__bf16)__builtin_nontemporal_load(xr + k) : (__bf16)0.f;
      }
#pragma unroll
      for (int i = 8; i < 16; ++i) a[i] = (__bf16)0.f;
      a1[7] = a;
    }

    // ---- encoder L1: 225->128, ELU (64 WMMA) ----
#pragma unroll
    for (int nt = 0; nt < 8; ++nt) {
      v8f acc = {};
      const __bf16* wrow = lw1 + (nt * 16 + nl) * S_W1;
#pragma unroll
      for (int kc = 0; kc < 8; ++kc)
        acc = WMMA_BF16(a1[kc], ld_bfrag(wrow + kc * 32 + grp * 8), acc);
      float bv = lbe1[nt * 16 + nl];
#pragma unroll
      for (int r = 0; r < 8; ++r)
        stg[(r + 8 * grp) * S_STG + nt * 16 + nl] = (__bf16)eluf(acc[r] + bv);
    }

    // ---- encoder L2: 128->64, ELU (16 WMMA) ----
    v16bf a2[4];
#pragma unroll
    for (int kc = 0; kc < 4; ++kc)
      a2[kc] = ld_afrag(stg + nl * S_STG + kc * 32 + grp * 8);
#pragma unroll
    for (int nt = 0; nt < 4; ++nt) {
      v8f acc = {};
      const __bf16* wrow = lw2 + (nt * 16 + nl) * S_W2;
#pragma unroll
      for (int kc = 0; kc < 4; ++kc)
        acc = WMMA_BF16(a2[kc], ld_bfrag(wrow + kc * 32 + grp * 8), acc);
      float bv = lbe2[nt * 16 + nl];
#pragma unroll
      for (int r = 0; r < 8; ++r)
        stg[(r + 8 * grp) * S_STG + nt * 16 + nl] = (__bf16)eluf(acc[r] + bv);
    }

    // ---- encoder L3: 64->16 (z) (2 WMMA); write z to global (NT) + LDS ----
    {
      v16bf a3a = ld_afrag(stg + nl * S_STG + 0 + grp * 8);
      v16bf a3b = ld_afrag(stg + nl * S_STG + 32 + grp * 8);
      v8f acc = {};
      const __bf16* wrow = lwmu + nl * S_WMU;
      acc = WMMA_BF16(a3a, ld_bfrag(wrow + 0 + grp * 8), acc);
      acc = WMMA_BF16(a3b, ld_bfrag(wrow + 32 + grp * 8), acc);
      float bv = lbmu[nl];
#pragma unroll
      for (int r = 0; r < 8; ++r) {
        int row = r + 8 * grp;
        float zv = acc[r] + bv;
        zstw[row * S_ZST + nl] = zv;
        __builtin_nontemporal_store(zv, o_z + (size_t)(R + row) * LATENT + nl);
      }
    }

    // ---- z_p = z @ w_p^T + b_p (per-lane, 4 dots each) ----
#pragma unroll
    for (int kk = 0; kk < 4; ++kk) {
      int k = grp * 4 + kk;
      float s = lbp[k];
#pragma unroll
      for (int j = 0; j < LATENT; ++j) s += zstw[nl * S_ZST + j] * lwp[k * LATENT + j];
      zpw[nl * S_ZP + k] = s;
    }

    // ---- quantizer GEMM: argmin_n ( ||e_p||^2 - 2 z_p . e_p ) (32 WMMA) ----
    v16bf aq;
#pragma unroll
    for (int i = 0; i < 16; ++i) aq[i] = (__bf16)0.f;
    if (grp == 0) {
#pragma unroll
      for (int i = 0; i < PROJ; ++i) aq[i] = (__bf16)zpw[nl * S_ZP + i];
    }
    float minv[8];
    int mini[8];
#pragma unroll
    for (int r = 0; r < 8; ++r) { minv[r] = __builtin_inff(); mini[r] = 0; }
    for (int nt = 0; nt < 32; ++nt) {
      int n = nt * 16 + nl;
      v8f dz = {};
      dz = WMMA_BF16(aq, ld_bfrag(lep + n * S_EP + grp * 8), dz);
      float esv = les[n];
#pragma unroll
      for (int r = 0; r < 8; ++r) {
        float v = dz[r] + esv;
        if (v < minv[r]) { minv[r] = v; mini[r] = n; }
      }
    }
    // butterfly argmin across the 16 lanes sharing each row (width 16)
#pragma unroll
    for (int r = 0; r < 8; ++r) {
#pragma unroll
      for (int off = 8; off > 0; off >>= 1) {
        float ov = __shfl_xor(minv[r], off, 16);
        int   oi = __shfl_xor(mini[r], off, 16);
        if (ov < minv[r] || (ov == minv[r] && oi < mini[r])) { minv[r] = ov; mini[r] = oi; }
      }
    }
    if (nl == 0) {
#pragma unroll
      for (int r = 0; r < 8; ++r) clw[8 * grp + r] = mini[r];
    }

    // ---- quantized gather -> global f32 (NT) + bf16 stage (K padded 16->32) ----
    {
      int row = lane >> 1, half = lane & 1;
      int cl = clw[row];
      const float* er = lemb + cl * LATENT;
      float* qrow = o_q + (size_t)(R + row) * LATENT;
      __builtin_nontemporal_store(*(const v4f*)(er + half * 8),     (v4f*)(qrow + half * 8));
      __builtin_nontemporal_store(*(const v4f*)(er + half * 8 + 4), (v4f*)(qrow + half * 8 + 4));
#pragma unroll
      for (int j = 0; j < 8; ++j) {
        int c = half * 8 + j;
        stg[row * S_STG + c] = (__bf16)er[c];
      }
      v4u z4 = {0u, 0u, 0u, 0u};
      *(v4u*)(stg + row * S_STG + 16 + half * 8) = z4;  // zero K pad 16..31
    }

    // ---- one_hot rows (dominant HBM traffic): NT zero stream + 1-lane patch ----
    // Same-wave global stores retire in order, so the scalar 1.0f patch lands
    // after the zero fill of its row.
    {
      v4f z4 = {0.f, 0.f, 0.f, 0.f};
#pragma unroll
      for (int row = 0; row < 16; ++row) {
        float* orow = o_oh + (size_t)(R + row) * NUM_EMB + lane * 16;
#pragma unroll
        for (int q = 0; q < 4; ++q)
          __builtin_nontemporal_store(z4, (v4f*)(orow + q * 4));
      }
      if (grp == 0) {
        int cl = clw[nl];
        __builtin_nontemporal_store(1.0f, o_oh + (size_t)(R + nl) * NUM_EMB + cl);
      }
    }

    // ---- decoder L1: 16->64, ELU (4 WMMA) ----
    {
      v16bf ad = ld_afrag(stg + nl * S_STG + grp * 8);
#pragma unroll
      for (int nt = 0; nt < 4; ++nt) {
        v8f acc = {};
        acc = WMMA_BF16(ad, ld_bfrag(lwd1 + (nt * 16 + nl) * S_WD1 + grp * 8), acc);
        float bv = lbd1[nt * 16 + nl];
#pragma unroll
        for (int r = 0; r < 8; ++r)
          stg[(r + 8 * grp) * S_STG + nt * 16 + nl] = (__bf16)eluf(acc[r] + bv);
      }
    }

    // ---- decoder L2: 64->128, ELU (16 WMMA) ----
    {
      v16bf ad[2];
#pragma unroll
      for (int kc = 0; kc < 2; ++kc)
        ad[kc] = ld_afrag(stg + nl * S_STG + kc * 32 + grp * 8);
#pragma unroll
      for (int nt = 0; nt < 8; ++nt) {
        v8f acc = {};
        const __bf16* wrow = lwd2 + (nt * 16 + nl) * S_WD2;
#pragma unroll
        for (int kc = 0; kc < 2; ++kc)
          acc = WMMA_BF16(ad[kc], ld_bfrag(wrow + kc * 32 + grp * 8), acc);
        float bv = lbd2[nt * 16 + nl];
#pragma unroll
        for (int r = 0; r < 8; ++r)
          stg[(r + 8 * grp) * S_STG + nt * 16 + nl] = (__bf16)eluf(acc[r] + bv);
      }
    }

    // ---- decoder L3: 128->45 (12 WMMA) -> x_hat (NT) ----
    {
      v16bf ad[4];
#pragma unroll
      for (int kc = 0; kc < 4; ++kc)
        ad[kc] = ld_afrag(stg + nl * S_STG + kc * 32 + grp * 8);
#pragma unroll
      for (int nt = 0; nt < 3; ++nt) {
        v8f acc = {};
        const __bf16* wrow = lwd3 + (nt * 16 + nl) * S_WD3;
#pragma unroll
        for (int kc = 0; kc < 4; ++kc)
          acc = WMMA_BF16(ad[kc], ld_bfrag(wrow + kc * 32 + grp * 8), acc);
        int col = nt * 16 + nl;
        float bv = lbd3[col];
        if (col < OUT_DIM) {
#pragma unroll
          for (int r = 0; r < 8; ++r) {
            int row = r + 8 * grp;
            __builtin_nontemporal_store(acc[r] + bv,
                                        o_xhat + (size_t)(R + row) * OUT_DIM + col);
          }
        }
      }
    }
  }
}

extern "C" void kernel_launch(void* const* d_in, const int* in_sizes, int n_in,
                              void* d_out, int out_size, void* d_ws, size_t ws_size,
                              hipStream_t stream) {
  (void)in_sizes; (void)n_in; (void)out_size; (void)d_ws; (void)ws_size;
  const float* x    = (const float*)d_in[0];
  const float* w_e1 = (const float*)d_in[1];
  const float* b_e1 = (const float*)d_in[2];
  const float* w_e2 = (const float*)d_in[3];
  const float* b_e2 = (const float*)d_in[4];
  const float* w_mu = (const float*)d_in[5];
  const float* b_mu = (const float*)d_in[6];
  const float* w_p  = (const float*)d_in[7];
  const float* b_p  = (const float*)d_in[8];
  const float* emb  = (const float*)d_in[9];
  const float* w_d1 = (const float*)d_in[10];
  const float* b_d1 = (const float*)d_in[11];
  const float* w_d2 = (const float*)d_in[12];
  const float* b_d2 = (const float*)d_in[13];
  const float* w_d3 = (const float*)d_in[14];
  const float* b_d3 = (const float*)d_in[15];

  float* out  = (float*)d_out;
  float* o_x  = out;
  float* o_q  = o_x + (size_t)B_TOT * OUT_DIM;
  float* o_z  = o_q + (size_t)B_TOT * LATENT;
  float* o_oh = o_z + (size_t)B_TOT * LATENT;

  (void)hipFuncSetAttribute((const void*)vqvae_fused,
                            hipFuncAttributeMaxDynamicSharedMemorySize,
                            (int)LDS_TOTAL);
  hipLaunchKernelGGL(vqvae_fused, dim3(GRID_BLKS), dim3(BLOCK_THREADS), LDS_TOTAL, stream,
                     x, w_e1, b_e1, w_e2, b_e2, w_mu, b_mu, w_p, b_p, emb,
                     w_d1, b_d1, w_d2, b_d2, w_d3, b_d3,
                     o_x, o_q, o_z, o_oh);
}
